// HSTUProvider_15375982920321
// MI455X (gfx1250) — compile-verified, matrix operand order checked by reading, not run on hardware
//
#include <hip/hip_runtime.h>

// ---------------- constants ----------------
static constexpr int B_ = 16, L_ = 1024, N_ = 1025;
static constexpr int HSTU_ = 256, H_ = 4, DQDV_ = 64;
static constexpr int NB_ = 4, NBUCK_ = 64;
static constexpr int PROV_V_ = 50000, DMODEL_ = 192, DT_DIM_ = 64, NRATE_ = 512;
static constexpr int M_ = B_ * N_;           // 16400 flattened rows (divisible by 16)
static constexpr int NP_ = 1056;             // padded seq length for attention tiles
static constexpr int UVQK_OUT_ = 1024;       // 2*H*(DV+DQ)

typedef __attribute__((ext_vector_type(16))) __bf16 v16bf;
typedef __attribute__((ext_vector_type(8)))  float  v8f;

// ---------------- device helpers ----------------
__device__ __forceinline__ float silu_f(float x) { return x / (1.0f + __expf(-x)); }

__device__ __forceinline__ v8f zero8() {
  v8f z;
#pragma unroll
  for (int i = 0; i < 8; ++i) z[i] = 0.0f;
  return z;
}

__device__ __forceinline__ v8f wmma_bf16(v16bf a, v16bf b, v8f c) {
  return __builtin_amdgcn_wmma_f32_16x16x32_bf16(false, a, false, b, (short)0, c,
                                                 false, false);
}

// Load a 16x32 bf16 tile (A layout, or B^T with identical striping) from a
// row-major matrix with leading dimension ld (elements).
// lane L: row = L&15 ; K halves: kb..kb+7 and 16+kb..16+kb+7 with kb = (L<16)?0:8
__device__ __forceinline__ v16bf ld_tile(const __bf16* p, int ld) {
  const int lane = threadIdx.x & 31;
  const int m    = lane & 15;
  const int kb   = (lane < 16) ? 0 : 8;
  const __bf16* r = p + (size_t)m * ld;
  v16bf out;
#pragma unroll
  for (int t = 0; t < 8; ++t) out[t] = r[kb + t];
#pragma unroll
  for (int t = 0; t < 8; ++t) out[8 + t] = r[16 + kb + t];
  return out;
}

// Same, but source is f32 (cast to bf16 on load) with per-row clamping.
__device__ __forceinline__ v16bf ld_tile_f32_clamp(const float* p, int ld,
                                                   int row0, int rowmax, int k0) {
  const int lane = threadIdx.x & 31;
  const int m    = lane & 15;
  const int kb   = (lane < 16) ? 0 : 8;
  int rr = row0 + m; if (rr > rowmax) rr = rowmax;
  const float* r = p + (size_t)rr * ld + k0;
  v16bf out;
#pragma unroll
  for (int t = 0; t < 8; ++t) out[t] = (__bf16)r[kb + t];
#pragma unroll
  for (int t = 0; t < 8; ++t) out[8 + t] = (__bf16)r[16 + kb + t];
  return out;
}

__device__ __forceinline__ float wave_sum(float v) {
#pragma unroll
  for (int o = 16; o > 0; o >>= 1) v += __shfl_xor(v, o, 32);
  return v;
}

// ---------------- one-time prep kernels ----------------
__global__ void k_prep_w(const float* __restrict__ uvqk, const float* __restrict__ ow,
                         __bf16* __restrict__ Wt, __bf16* __restrict__ oWt) {
  const int n1 = NB_ * HSTU_ * UVQK_OUT_;
  const int n2 = NB_ * HSTU_ * HSTU_;
  int idx = blockIdx.x * 256 + threadIdx.x;
  if (idx < n1) {
    int blk = idx / (HSTU_ * UVQK_OUT_);
    int r   = idx - blk * (HSTU_ * UVQK_OUT_);
    int o   = r / HSTU_, i = r - o * HSTU_;
    Wt[idx] = (__bf16)uvqk[(size_t)blk * HSTU_ * UVQK_OUT_ + (size_t)i * UVQK_OUT_ + o];
  } else if (idx < n1 + n2) {
    int j   = idx - n1;
    int blk = j / (HSTU_ * HSTU_);
    int r   = j - blk * (HSTU_ * HSTU_);
    int o   = r / HSTU_, i = r - o * HSTU_;
    oWt[j] = (__bf16)ow[(size_t)blk * HSTU_ * HSTU_ + (size_t)i * HSTU_ + o];
  }
}

__global__ void k_build_x(const int* __restrict__ seq, const int* __restrict__ delta_t,
                          const int* __restrict__ trigger,
                          const float* __restrict__ prov, const float* __restrict__ spec,
                          const float* __restrict__ dxe, const float* __restrict__ dte,
                          const float* __restrict__ pose,
                          float* __restrict__ x, int* __restrict__ ts,
                          unsigned char* __restrict__ pm) {
  const int n = blockIdx.x, b = blockIdx.y, c = threadIdx.x;
  float val;
  int padf = 0, tsv = 0;
  if (n < L_) {
    int s0 = seq[(b * L_ + n) * 2 + 0];
    int s1 = seq[(b * L_ + n) * 2 + 1];
    int dt = delta_t[b * L_ + n];
    dt = dt < 0 ? 0 : (dt > NRATE_ - 1 ? NRATE_ - 1 : dt);
    tsv = dt;
    padf = (s0 == 0);
    val = (c < DMODEL_)
            ? prov[(size_t)s0 * DMODEL_ + c] + spec[(size_t)s1 * DMODEL_ + c]
            : dte[(size_t)dt * DT_DIM_ + (c - DMODEL_)];
  } else {
    int tg = trigger[b];
    val = (c < DMODEL_) ? dxe[(size_t)tg * DMODEL_ + c] : 0.0f;
  }
  val = val * 16.0f + pose[(size_t)n * HSTU_ + c];   // sqrt(HSTU)=16
  if (padf) val = 0.0f;
  x[((size_t)(b * N_ + n)) * HSTU_ + c] = val;
  if (c == 0) { ts[b * N_ + n] = tsv; pm[b * N_ + n] = (unsigned char)padf; }
}

__global__ void k_bucket(const int* __restrict__ ts, unsigned char* __restrict__ bk) {
  const int b = blockIdx.y;
  int idx = blockIdx.x * 256 + threadIdx.x;
  if (idx >= N_ * N_) return;
  int i = idx / N_, j = idx - i * N_;
  const int* t = ts + b * N_;
  int ti1 = (i + 1 < N_) ? t[i + 1] : t[N_ - 1];
  int d = ti1 - t[j];
  float a = fmaxf(fabsf((float)d), 1.0f);
  int v = (int)(logf(a) * (1.0f / 0.301f));
  v = v < 0 ? 0 : (v > NBUCK_ ? NBUCK_ : v);
  bk[(size_t)b * N_ * N_ + idx] = (unsigned char)v;
}

// ---------------- layernorm kernels (one wave per 256-wide row) ----------------
__global__ void k_ln_cast(const float* __restrict__ src, const float* __restrict__ g,
                          const float* __restrict__ bb, __bf16* __restrict__ dst) {
  const int row = blockIdx.x, lane = threadIdx.x;
  const float* s = src + (size_t)row * HSTU_;
  float v[8], sum = 0.0f;
#pragma unroll
  for (int t = 0; t < 8; ++t) { v[t] = s[lane + 32 * t]; sum += v[t]; }
  sum = wave_sum(sum);
  float mean = sum * (1.0f / HSTU_), vs = 0.0f;
#pragma unroll
  for (int t = 0; t < 8; ++t) { float d = v[t] - mean; vs += d * d; }
  vs = wave_sum(vs);
  float rs = rsqrtf(vs * (1.0f / HSTU_) + 1e-5f);
  __bf16* d = dst + (size_t)row * HSTU_;
#pragma unroll
  for (int t = 0; t < 8; ++t) {
    int c = lane + 32 * t;
    d[c] = (__bf16)((v[t] - mean) * rs * g[c] + bb[c]);
  }
}

__global__ void k_ln_mul_cast(const float* __restrict__ src, const float* __restrict__ um,
                              const float* __restrict__ g, const float* __restrict__ bb,
                              __bf16* __restrict__ dst) {
  const int row = blockIdx.x, lane = threadIdx.x;
  const float* s = src + (size_t)row * HSTU_;
  const float* uu = um + (size_t)row * HSTU_;
  float v[8], sum = 0.0f;
#pragma unroll
  for (int t = 0; t < 8; ++t) { v[t] = s[lane + 32 * t]; sum += v[t]; }
  sum = wave_sum(sum);
  float mean = sum * (1.0f / HSTU_), vs = 0.0f;
#pragma unroll
  for (int t = 0; t < 8; ++t) { float d = v[t] - mean; vs += d * d; }
  vs = wave_sum(vs);
  float rs = rsqrtf(vs * (1.0f / HSTU_) + 1e-5f);
  __bf16* d = dst + (size_t)row * HSTU_;
#pragma unroll
  for (int t = 0; t < 8; ++t) {
    int c = lane + 32 * t;
    d[c] = (__bf16)(((v[t] - mean) * rs * g[c] + bb[c]) * uu[c]);
  }
}

__global__ void k_final_ln(const float* __restrict__ x, const float* __restrict__ g,
                           const float* __restrict__ bb, float* __restrict__ out_u,
                           __bf16* __restrict__ ub) {
  const int b = blockIdx.x, lane = threadIdx.x;
  const float* s = x + ((size_t)(b * N_ + (N_ - 1))) * HSTU_;
  float v[8], sum = 0.0f;
#pragma unroll
  for (int t = 0; t < 8; ++t) { v[t] = s[lane + 32 * t]; sum += v[t]; }
  sum = wave_sum(sum);
  float mean = sum * (1.0f / HSTU_), vs = 0.0f;
#pragma unroll
  for (int t = 0; t < 8; ++t) { float d = v[t] - mean; vs += d * d; }
  vs = wave_sum(vs);
  float rs = rsqrtf(vs * (1.0f / HSTU_) + 1e-5f);
#pragma unroll
  for (int t = 0; t < 8; ++t) {
    int c = lane + 32 * t;
    float r = (v[t] - mean) * rs * g[c] + bb[c];
    out_u[(size_t)b * HSTU_ + c] = r;
    if (c < DMODEL_) ub[(size_t)b * DMODEL_ + c] = (__bf16)r;
  }
}

// ---------------- WMMA GEMM: out = silu(xn @ uvqk), scatter u/q/k/vT ----------------
__global__ void k_gemm_uvqk(const __bf16* __restrict__ xn, const __bf16* __restrict__ Wt,
                            float* __restrict__ u, __bf16* __restrict__ qb,
                            __bf16* __restrict__ kb2, __bf16* __restrict__ vT) {
  const int mt = blockIdx.x, g = blockIdx.y;        // 16 rows x 64 cols per wave
  const __bf16* A  = xn + (size_t)mt * 16 * HSTU_;
  const __bf16* Wb = Wt + (size_t)g * 64 * HSTU_;
  v8f acc[4];
#pragma unroll
  for (int ct = 0; ct < 4; ++ct) acc[ct] = zero8();
#pragma unroll
  for (int kk = 0; kk < 8; ++kk) {
    v16bf va = ld_tile(A + kk * 32, HSTU_);
#pragma unroll
    for (int ct = 0; ct < 4; ++ct) {
      v16bf vb = ld_tile(Wb + (size_t)ct * 16 * HSTU_ + kk * 32, HSTU_);
      acc[ct] = wmma_bf16(va, vb, acc[ct]);
    }
  }
  const int lane = threadIdx.x & 31;
  const int mrow = (lane >> 4) << 3, ncol = lane & 15;
#pragma unroll
  for (int ct = 0; ct < 4; ++ct) {
#pragma unroll
    for (int r = 0; r < 8; ++r) {
      int row = mt * 16 + mrow + r;
      int c   = g * 64 + ct * 16 + ncol;
      float val = silu_f(acc[ct][r]);
      int bb = row / N_, n = row - bb * N_;
      if (c < 256) {
        u[(size_t)row * HSTU_ + c] = val;
      } else if (c < 512) {
        int h = (c - 256) >> 6, dd = (c - 256) & 63;
        vT[(((size_t)(bb * H_ + h)) * 64 + dd) * NP_ + n] = (__bf16)val;
      } else if (c < 768) {
        int h = (c - 512) >> 6, dd = (c - 512) & 63;
        qb[(((size_t)(bb * H_ + h)) * NP_ + n) * 64 + dd] = (__bf16)val;
      } else {
        int h = (c - 768) >> 6, dd = (c - 768) & 63;
        kb2[(((size_t)(bb * H_ + h)) * NP_ + n) * 64 + dd] = (__bf16)val;
      }
    }
  }
}

// ---------------- attention: one wave per (b, h, 16-row q-tile) ----------------
__global__ void k_attn(const __bf16* __restrict__ qb, const __bf16* __restrict__ kb2,
                       const __bf16* __restrict__ vT, const unsigned char* __restrict__ bk,
                       const unsigned char* __restrict__ pm,
                       const float* __restrict__ posw, const float* __restrict__ tsw,
                       float* __restrict__ ao) {
  __shared__ float spos[2 * N_ - 1];
  __shared__ float stw[NBUCK_ + 1];
  __shared__ unsigned char spm[N_];
  __shared__ __bf16 sS[16 * 32];

  const int it = blockIdx.x, h = blockIdx.y, b = blockIdx.z;
  const int tid = threadIdx.x;
  for (int i = tid; i < 2 * N_ - 1; i += 32) spos[i] = posw[i];
  for (int i = tid; i < NBUCK_ + 1; i += 32) stw[i] = tsw[i];
  for (int i = tid; i < N_; i += 32) spm[i] = pm[b * N_ + i];
  __syncthreads();

  const int i0 = it * 16;
  const __bf16* qp = qb + ((size_t)(b * H_ + h) * NP_ + i0) * 64;
  v16bf qa0 = ld_tile(qp, 64);
  v16bf qa1 = ld_tile(qp + 32, 64);

  v8f o[4];
#pragma unroll
  for (int d = 0; d < 4; ++d) o[d] = zero8();

  const int lane = tid & 31;
  const int mrow = (lane >> 4) << 3, ncol = lane & 15;
  const float invN = 1.0f / (float)N_;
  const unsigned char* bkb = bk + (size_t)b * N_ * N_;
  const int jmax = (i0 + 15 < N_ - 1) ? (i0 + 15) : (N_ - 1);

  for (int j0 = 0; j0 <= jmax; j0 += 32) {
    const __bf16* kp = kb2 + ((size_t)(b * H_ + h) * NP_ + j0) * 64;
    v8f s0 = zero8(), s1 = zero8();
    s0 = wmma_bf16(qa0, ld_tile(kp, 64), s0);
    s0 = wmma_bf16(qa1, ld_tile(kp + 32, 64), s0);
    s1 = wmma_bf16(qa0, ld_tile(kp + 16 * 64, 64), s1);
    s1 = wmma_bf16(qa1, ld_tile(kp + 16 * 64 + 32, 64), s1);

#pragma unroll
    for (int r = 0; r < 8; ++r) {
      int ig = i0 + mrow + r;
      float v0 = 0.0f, v1 = 0.0f;
      if (ig < N_) {
        const unsigned char* bkrow = bkb + (size_t)ig * N_;
        int jg0 = j0 + ncol;
        if (jg0 <= ig && !spm[jg0])
          v0 = silu_f(s0[r] + spos[(N_ - 1) + jg0 - ig] + stw[bkrow[jg0]]) * invN;
        int jg1 = jg0 + 16;
        if (jg1 <= ig && !spm[jg1])
          v1 = silu_f(s1[r] + spos[(N_ - 1) + jg1 - ig] + stw[bkrow[jg1]]) * invN;
      }
      sS[(mrow + r) * 32 + ncol]      = (__bf16)v0;
      sS[(mrow + r) * 32 + 16 + ncol] = (__bf16)v1;
    }
    __syncthreads();
    v16bf pa = ld_tile(sS, 32);
    const __bf16* vbase = vT + ((size_t)(b * H_ + h) * 64) * NP_ + j0;
#pragma unroll
    for (int d = 0; d < 4; ++d) {
      v16bf vb = ld_tile(vbase + (size_t)d * 16 * NP_, NP_);
      o[d] = wmma_bf16(pa, vb, o[d]);
    }
    __syncthreads();
  }

#pragma unroll
  for (int d = 0; d < 4; ++d) {
#pragma unroll
    for (int r = 0; r < 8; ++r) {
      int ig = i0 + mrow + r;
      if (ig < N_)
        ao[((size_t)(b * N_ + ig)) * HSTU_ + h * 64 + d * 16 + ncol] = o[d][r];
    }
  }
}

// ---------------- output GEMM: x = (u*ln(ao)) @ o_w + o_b + x, masked ----------------
__global__ void k_gemm_out(const __bf16* __restrict__ yin, const __bf16* __restrict__ oWt,
                           const float* __restrict__ ob, const unsigned char* __restrict__ pm,
                           float* __restrict__ x) {
  const int mt = blockIdx.x, g = blockIdx.y;
  const __bf16* A  = yin + (size_t)mt * 16 * HSTU_;
  const __bf16* Wb = oWt + (size_t)g * 64 * HSTU_;
  v8f acc[4];
#pragma unroll
  for (int ct = 0; ct < 4; ++ct) acc[ct] = zero8();
#pragma unroll
  for (int kk = 0; kk < 8; ++kk) {
    v16bf va = ld_tile(A + kk * 32, HSTU_);
#pragma unroll
    for (int ct = 0; ct < 4; ++ct) {
      v16bf vb = ld_tile(Wb + (size_t)ct * 16 * HSTU_ + kk * 32, HSTU_);
      acc[ct] = wmma_bf16(va, vb, acc[ct]);
    }
  }
  const int lane = threadIdx.x & 31;
  const int mrow = (lane >> 4) << 3, ncol = lane & 15;
#pragma unroll
  for (int ct = 0; ct < 4; ++ct) {
#pragma unroll
    for (int r = 0; r < 8; ++r) {
      int row = mt * 16 + mrow + r;
      int c   = g * 64 + ct * 16 + ncol;
      float val = acc[ct][r] + ob[c] + x[(size_t)row * HSTU_ + c];
      if (pm[row]) val = 0.0f;
      x[(size_t)row * HSTU_ + c] = val;
    }
  }
}

// ---------------- logits: (16 x 49998) = u[:, :192] @ provider_emb[2:].T ----------------
__global__ void k_logits(const __bf16* __restrict__ ub, const float* __restrict__ prov,
                         float* __restrict__ out) {
  const int c0 = blockIdx.x * 16;
  v8f acc = zero8();
#pragma unroll
  for (int kk = 0; kk < 6; ++kk) {
    v16bf va = ld_tile(ub + kk * 32, DMODEL_);
    v16bf vb = ld_tile_f32_clamp(prov, DMODEL_, 2 + c0, PROV_V_ - 1, kk * 32);
    acc = wmma_bf16(va, vb, acc);
  }
  const int lane = threadIdx.x & 31;
  const int mrow = (lane >> 4) << 3, ncol = lane & 15;
  const int ncols = PROV_V_ - 2;
#pragma unroll
  for (int r = 0; r < 8; ++r) {
    int m = mrow + r;
    int col = c0 + ncol;
    if (col < ncols) out[(size_t)m * ncols + col] = acc[r];
  }
}

// ---------------- host launcher ----------------
extern "C" void kernel_launch(void* const* d_in, const int* in_sizes, int n_in,
                              void* d_out, int out_size, void* d_ws, size_t ws_size,
                              hipStream_t stream) {
  (void)in_sizes; (void)n_in; (void)out_size; (void)ws_size;
  const int*   seq     = (const int*)d_in[0];
  const int*   delta_t = (const int*)d_in[1];
  const int*   trigger = (const int*)d_in[2];
  /* d_in[3] seq_len unused by reference */
  const float* prov    = (const float*)d_in[4];
  const float* spec    = (const float*)d_in[5];
  const float* dxe     = (const float*)d_in[6];
  const float* dte     = (const float*)d_in[7];
  const float* pose    = (const float*)d_in[8];
  const float* uvqk    = (const float*)d_in[9];
  const float* ow      = (const float*)d_in[10];
  const float* obias   = (const float*)d_in[11];
  const float* nxg     = (const float*)d_in[12];
  const float* nxb     = (const float*)d_in[13];
  const float* nag     = (const float*)d_in[14];
  const float* nab     = (const float*)d_in[15];
  const float* tsw     = (const float*)d_in[16];
  const float* posw    = (const float*)d_in[17];
  const float* ong     = (const float*)d_in[18];
  const float* onb     = (const float*)d_in[19];

  char* ws = (char*)d_ws;
  size_t off = 0;
  auto take = [&](size_t bytes) -> char* {
    char* p = ws + off;
    off = (off + bytes + 255) & ~(size_t)255;
    return p;
  };
  float*  x    = (float*)take((size_t)M_ * HSTU_ * 4);
  __bf16* xn   = (__bf16*)take((size_t)M_ * HSTU_ * 2);
  float*  u    = (float*)take((size_t)M_ * HSTU_ * 4);
  float*  ao   = (float*)take((size_t)M_ * HSTU_ * 4);
  __bf16* yin  = (__bf16*)take((size_t)M_ * HSTU_ * 2);
  const size_t qsz = (size_t)B_ * H_ * NP_ * 64 * 2;   // multiple of 256
  __bf16* qb   = (__bf16*)take(qsz);
  __bf16* kb2  = (__bf16*)take(qsz);
  __bf16* vT   = (__bf16*)take(qsz);
  __bf16* Wt   = (__bf16*)take((size_t)NB_ * HSTU_ * UVQK_OUT_ * 2);
  __bf16* oWt  = (__bf16*)take((size_t)NB_ * HSTU_ * HSTU_ * 2);
  unsigned char* bk = (unsigned char*)take((size_t)B_ * N_ * N_);
  unsigned char* pm = (unsigned char*)take((size_t)M_);
  int*    ts   = (int*)take((size_t)M_ * 4);
  __bf16* ub   = (__bf16*)take((size_t)B_ * DMODEL_ * 2);

  float* out_logits = (float*)d_out;
  float* out_u      = out_logits + (size_t)B_ * (PROV_V_ - 2);

  // one-time prep
  const int prep_total = NB_ * HSTU_ * UVQK_OUT_ + NB_ * HSTU_ * HSTU_;
  k_prep_w<<<(prep_total + 255) / 256, 256, 0, stream>>>(uvqk, ow, Wt, oWt);
  hipMemsetAsync(qb, 0, qsz * 3, stream);   // zero q/k/vT pads (contiguous)
  k_build_x<<<dim3(N_, B_), HSTU_, 0, stream>>>(seq, delta_t, trigger, prov, spec,
                                                dxe, dte, pose, x, ts, pm);
  k_bucket<<<dim3((N_ * N_ + 255) / 256, B_), 256, 0, stream>>>(ts, bk);

  for (int blk = 0; blk < NB_; ++blk) {
    k_ln_cast<<<M_, 32, 0, stream>>>(x, nxg + blk * HSTU_, nxb + blk * HSTU_, xn);
    k_gemm_uvqk<<<dim3(M_ / 16, 16), 32, 0, stream>>>(
        xn, Wt + (size_t)blk * HSTU_ * UVQK_OUT_, u, qb, kb2, vT);
    k_attn<<<dim3((N_ + 15) / 16, H_, B_), 32, 0, stream>>>(
        qb, kb2, vT, bk, pm, posw + blk * (2 * N_ - 1), tsw + blk * (NBUCK_ + 1), ao);
    k_ln_mul_cast<<<M_, 32, 0, stream>>>(ao, u, nag + blk * HSTU_, nab + blk * HSTU_, yin);
    k_gemm_out<<<dim3(M_ / 16, 4), 32, 0, stream>>>(
        yin, oWt + (size_t)blk * HSTU_ * HSTU_, obias + blk * HSTU_, pm, x);
  }

  k_final_ln<<<B_, 32, 0, stream>>>(x, ong, onb, out_u, ub);
  k_logits<<<((PROV_V_ - 2) + 15) / 16, 32, 0, stream>>>(ub, prov, out_logits);
}
